// SeNaTraStage_29111288332440
// MI455X (gfx1250) — matrix operand order repaired and checked by reading, not compiled, not run on hardware
//
#include <hip/hip_runtime.h>
#include <hip/hip_bf16.h>
#include <math.h>

// ---------------------------------------------------------------------------
// SeNaTra stage for gfx1250 (CDNA5, wave32, WMMA).
// GEMMs use v_wmma_f32_16x16x32_bf16 (bf16 in, f32 accum), 64x128 tiles,
// templated epilogues. NA attention, LN, RoPE, softmax are wave32 VALU kernels.
// ---------------------------------------------------------------------------

typedef __attribute__((ext_vector_type(16))) __bf16 v16bf;
typedef __attribute__((ext_vector_type(8)))  float  v8f;

union Frag { unsigned int u[8]; v16bf v; };

__device__ __forceinline__ unsigned short f2bf(float f) {
  unsigned int u = __float_as_uint(f);
  unsigned int r = ((u >> 16) & 1u) + 0x7FFFu;   // round-to-nearest-even
  return (unsigned short)((u + r) >> 16);
}
__device__ __forceinline__ unsigned int f2bf2(float a, float b) {
  return (unsigned int)f2bf(a) | ((unsigned int)f2bf(b) << 16);
}

__device__ __forceinline__ float wave_sum(float v) {
#pragma unroll
  for (int o = 16; o > 0; o >>= 1) v += __shfl_xor(v, o, 32);
  return v;
}
__device__ __forceinline__ float wave_max(float v) {
#pragma unroll
  for (int o = 16; o > 0; o >>= 1) v = fmaxf(v, __shfl_xor(v, o, 32));
  return v;
}
__device__ __forceinline__ float gelu_exact(float x) {
  return 0.5f * x * (1.0f + erff(x * 0.70710678118654752440f));
}

#define GF_BIAS  1
#define GF_GELU  2
#define GF_RESID 4
#define GF_PERM  8

// ---------------------------------------------------------------------------
// WMMA GEMM: C[M,N] = op(A[M,K] @ W[N,K]^T + bias)
// Workgroup tile 64x128, 8 waves (4 M-sub x 2 N-halves), 4 acc/wave.
// M % 64 == 0, N % 128 == 0, K % 32 == 0 (true for every call here).
// FLAGS is compile-time -> straight-line epilogue per variant.
// ---------------------------------------------------------------------------
template <int FLAGS>
__global__ __launch_bounds__(256) void gemm_wmma(
    const float* __restrict__ A, int lda,
    const float* __restrict__ W, int ldw,
    const float* __restrict__ bias,
    float* __restrict__ C, int ldc,
    int K)
{
  __shared__ unsigned short la[64 * 36];    // A tile [m][k], padded stride
  __shared__ unsigned short lw[32 * 132];   // W tile transposed [k][n], padded
  const int tid  = threadIdx.x;
  const int wv   = tid >> 5, lane = tid & 31;
  const int wm   = wv & 3,   wn   = wv >> 2;
  const int m0   = blockIdx.y << 6, n0 = blockIdx.x << 7;
  const int hh   = lane >> 4, lr = lane & 15;
  const v8f vz = {0.f,0.f,0.f,0.f,0.f,0.f,0.f,0.f};
  v8f acc[4] = {vz, vz, vz, vz};

  const int aidx = tid << 3;                 // A fill: 8 elems / thread
  const int arow = aidx >> 5, acol = aidx & 31;
  const int widx = tid << 4;                 // W fill: 16 elems / thread
  const int wrow = widx >> 5, wcol = widx & 31;
  const float* aptr = A + (size_t)(m0 + arow) * lda + acol;
  const float* wptr = W + (size_t)(n0 + wrow) * ldw + wcol;

  for (int k0 = 0; k0 < K; k0 += 32) {
    __syncthreads();
    {   // A tile: f32 -> bf16, packed u32 LDS stores
      const float* src = aptr + k0;
      unsigned int* dst = (unsigned int*)&la[arow * 36 + acol];
#pragma unroll
      for (int e = 0; e < 4; ++e) dst[e] = f2bf2(src[2 * e], src[2 * e + 1]);
    }
    {   // W tile: stored transposed [k][n] (16-bit scattered stores)
      const float* src = wptr + k0;
#pragma unroll
      for (int e = 0; e < 16; ++e) lw[(wcol + e) * 132 + wrow] = f2bf(src[e]);
    }
    if (k0 + 32 < K) {   // prefetch next K-step (global_prefetch_b8)
      __builtin_prefetch(aptr + k0 + 32, 0, 3);
      __builtin_prefetch(wptr + k0 + 32, 0, 3);
    }
    __syncthreads();
    // A fragment: lane<16 -> M=lr, K pairs {0..7,16..23}; lane>=16 -> {8..15,24..31}
    Frag fa;
    const unsigned int* pa = (const unsigned int*)&la[(wm * 16 + lr) * 36];
    const int ko = hh * 4;
    fa.u[0] = pa[ko + 0];  fa.u[1] = pa[ko + 1];
    fa.u[2] = pa[ko + 2];  fa.u[3] = pa[ko + 3];
    fa.u[4] = pa[ko + 8];  fa.u[5] = pa[ko + 9];
    fa.u[6] = pa[ko + 10]; fa.u[7] = pa[ko + 11];
    // B fragments: lane%16 = N column, VGPR r = K pair in same order as A
    const unsigned short* pb = &lw[wn * 64 + lr];
#pragma unroll
    for (int s = 0; s < 4; ++s) {
      Frag fb;
      const int nb2 = s * 16;
#pragma unroll
      for (int r = 0; r < 8; ++r) { const int kk = hh * 8 + 2 * r + ((r >> 2) << 3); fb.u[r] = (unsigned int)pb[kk * 132 + nb2] | ((unsigned int)pb[(kk + 1) * 132 + nb2] << 16); }
      acc[s] = __builtin_amdgcn_wmma_f32_16x16x32_bf16(false, fa.v, false, fb.v,
                                                       (short)0, acc[s], false, false);
    }
  }
  // Epilogue: C/D layout -> VGPR r: M = r + 8*(lane>=16); N = lane%16
  const int mbase = m0 + wm * 16 + hh * 8;
#pragma unroll
  for (int s = 0; s < 4; ++s) {
    const int ncol = n0 + wn * 64 + s * 16 + lr;
    const float bv = (FLAGS & GF_BIAS) ? bias[ncol] : 0.f;
#pragma unroll
    for (int r = 0; r < 8; ++r) {
      const int m = mbase + r;
      float v = acc[s][r] + bv;
      if (FLAGS & GF_GELU) v = gelu_exact(v);
      size_t orow = (size_t)m;
      if (FLAGS & GF_PERM) {   // to_groups row permutation
        const int b_ = m / 3136, rem = m % 3136;
        const int hi = rem / 56, wi = rem % 56;
        orow = (size_t)((b_ * 4 + (hi & 1) * 2 + (wi & 1)) * 784 + (hi >> 1) * 28 + (wi >> 1));
      }
      float* cp = C + orow * ldc + ncol;
      if (FLAGS & GF_RESID) v += *cp;
      *cp = v;
    }
  }
}

// ---------------------------------------------------------------------------
// Seed conv 3x3 stride 2 pad 1, D=256 -> 256, as im2col WMMA GEMM.
// M = 3136 (b,ho,wo), N = 256 (64-wide tiles), K = 2304 (tap-major).
// Weight element (n, k) = g_seed_w[n*2304 + c*9 + tap].
// ---------------------------------------------------------------------------
__global__ __launch_bounds__(256) void conv_seed_wmma(
    const float* __restrict__ X, const float* __restrict__ Wt,
    float* __restrict__ C)
{
  __shared__ unsigned short la[64 * 36];
  __shared__ unsigned short lw[32 * 68];
  const int tid  = threadIdx.x;
  const int wv   = tid >> 5, lane = tid & 31;
  const int wm   = wv & 3,   wn   = wv >> 2;
  const int m0   = blockIdx.y << 6, n0 = blockIdx.x << 6;
  const int hh   = lane >> 4, lr = lane & 15;
  v8f acc0 = {0.f,0.f,0.f,0.f,0.f,0.f,0.f,0.f};
  v8f acc1 = acc0;
  const int idx  = tid << 3;
  const int arow = idx >> 5, acol = idx & 31;
  const int m    = m0 + arow;
  const int bb   = m / 784, rem = m % 784;
  const int ho   = rem / 28, wo = rem % 28;

  for (int k0 = 0; k0 < 2304; k0 += 32) {
    const int tap = k0 >> 8, c0 = k0 & 255;
    const int kh = tap / 3, kw = tap % 3;
    __syncthreads();
    {   // gathered A tile (im2col with zero padding), packed u32 stores
      const int hi = ho * 2 - 1 + kh, wi = wo * 2 - 1 + kw;
      unsigned int* dst = (unsigned int*)&la[arow * 36 + acol];
      if (hi >= 0 && hi < 56 && wi >= 0 && wi < 56) {
        const float* src = X + ((size_t)(bb * 3136 + hi * 56 + wi)) * 256 + c0 + acol;
#pragma unroll
        for (int e = 0; e < 4; ++e) dst[e] = f2bf2(src[2 * e], src[2 * e + 1]);
      } else {
#pragma unroll
        for (int e = 0; e < 4; ++e) dst[e] = 0u;
      }
    }
    {   // gathered W tile, transposed to [k][n]
      const float* src = Wt + (size_t)(n0 + arow) * 2304 + (size_t)(c0 + acol) * 9 + tap;
#pragma unroll
      for (int e = 0; e < 8; ++e) lw[(acol + e) * 68 + arow] = f2bf(src[e * 9]);
    }
    __syncthreads();
    Frag fa, fb0, fb1;
    const unsigned int* pa = (const unsigned int*)&la[(wm * 16 + lr) * 36];
    const int ko = hh * 4;
    fa.u[0] = pa[ko + 0];  fa.u[1] = pa[ko + 1];
    fa.u[2] = pa[ko + 2];  fa.u[3] = pa[ko + 3];
    fa.u[4] = pa[ko + 8];  fa.u[5] = pa[ko + 9];
    fa.u[6] = pa[ko + 10]; fa.u[7] = pa[ko + 11];
    const unsigned short* pb = &lw[wn * 32 + lr];
    const int nb2 = 0;
#pragma unroll
    for (int r = 0; r < 8; ++r) { const int kk = hh * 8 + 2 * r + ((r >> 2) << 3); fb0.u[r] = (unsigned int)pb[kk * 68 + nb2] | ((unsigned int)pb[(kk + 1) * 68 + nb2] << 16); fb1.u[r] = (unsigned int)pb[kk * 68 + nb2 + 16] | ((unsigned int)pb[(kk + 1) * 68 + nb2 + 16] << 16); }
    acc0 = __builtin_amdgcn_wmma_f32_16x16x32_bf16(false, fa.v, false, fb0.v,
                                                   (short)0, acc0, false, false);
    acc1 = __builtin_amdgcn_wmma_f32_16x16x32_bf16(false, fa.v, false, fb1.v,
                                                   (short)0, acc1, false, false);
  }
  const int ncol0 = n0 + wn * 32 + lr;
  const int ncol1 = ncol0 + 16;
  const int mbase = m0 + wm * 16 + hh * 8;
#pragma unroll
  for (int r = 0; r < 8; ++r) {
    float* cp = C + (size_t)(mbase + r) * 256;
    cp[ncol0] = acc0[r];
    cp[ncol1] = acc1[r];
  }
}

// ---------------------------------------------------------------------------
// LayerNorm over last dim 256; one wave per row. mode 0: Y = ln; 1: Y += ln.
// ---------------------------------------------------------------------------
__global__ __launch_bounds__(256) void ln256(
    const float* __restrict__ X, const float* __restrict__ g,
    const float* __restrict__ b, float* __restrict__ Y, int mode)
{
  const int wv = threadIdx.x >> 5, lane = threadIdx.x & 31;
  const size_t row = (size_t)blockIdx.x * 8 + wv;
  const float* x = X + row * 256;
  float vals[8];
  float s = 0.f;
#pragma unroll
  for (int j = 0; j < 8; ++j) { vals[j] = x[j * 32 + lane]; s += vals[j]; }
  s = wave_sum(s);
  const float mean = s * (1.f / 256.f);
  float q = 0.f;
#pragma unroll
  for (int j = 0; j < 8; ++j) { const float d = vals[j] - mean; q += d * d; }
  q = wave_sum(q);
  const float rstd = rsqrtf(q * (1.f / 256.f) + 1e-5f);
  float* y = Y + row * 256;
#pragma unroll
  for (int j = 0; j < 8; ++j) {
    const int d = j * 32 + lane;
    const float o = (vals[j] - mean) * rstd * g[d] + b[d];
    if (mode) y[d] += o; else y[d] = o;
  }
}

__global__ void zero_f32(float* __restrict__ p, int n) {
  const int i = blockIdx.x * 256 + threadIdx.x;
  if (i < n) p[i] = 0.f;
}

// ---------------------------------------------------------------------------
// Group NA (3x3).  Reference swap: attn = na2d_qk(k, broadcast(q)),
// i.e. center vector = k[b,g,i,j], gathered neighbors = q[b, ih, iw].
// One wave per (b,g,i,j). attn = softmax((k.q + rpb)*exp(tau)) + 1e-6.
// ---------------------------------------------------------------------------
__global__ __launch_bounds__(256) void na_group_qk(
    const float* __restrict__ Q, const float* __restrict__ Kg,
    const float* __restrict__ rpb, const float* __restrict__ tau,
    float* __restrict__ attn)
{
  const int wv = threadIdx.x >> 5, lane = threadIdx.x & 31;
  const int gid = blockIdx.x * 8 + wv;            // (b*4+g)*784 + i*28 + j
  const int bg = gid / 784, pos = gid % 784;
  const int bb = bg >> 2, g = bg & 3;
  const int i = pos / 28, j = pos % 28;
  const int si = min(max(i - 1, 0), 25), sj = min(max(j - 1, 0), 25);
  const float* kc = Kg + ((size_t)bg * 784 + pos) * 256 + lane * 8;
  float k8[8];
#pragma unroll
  for (int e = 0; e < 8; ++e) k8[e] = kc[e];
  float logit[9];
#pragma unroll
  for (int a = 0; a < 3; ++a)
#pragma unroll
    for (int c = 0; c < 3; ++c) {
      const float* qp = Q + ((size_t)bb * 784 + (si + a) * 28 + (sj + c)) * 256 + lane * 8;
      float acc = 0.f;
#pragma unroll
      for (int e = 0; e < 8; ++e) acc += k8[e] * qp[e];
      logit[a * 3 + c] = wave_sum(acc);
    }
  const float sg = expf(tau[0]);
  float mx = -1e30f;
#pragma unroll
  for (int t = 0; t < 9; ++t) { logit[t] = (logit[t] + rpb[g * 9 + t]) * sg; mx = fmaxf(mx, logit[t]); }
  float ssum = 0.f;
#pragma unroll
  for (int t = 0; t < 9; ++t) { logit[t] = expf(logit[t] - mx); ssum += logit[t]; }
  const float inv = 1.f / ssum;
  if (lane == 0) {
    float* ap = attn + (size_t)gid * 9;
#pragma unroll
    for (int t = 0; t < 9; ++t) ap[t] = logit[t] * inv + 1e-6f;
  }
}

// Column scatter: col[b, p] += attn over all (g,i,j,a,c) mapping to p.
__global__ void na_group_scatter(const float* __restrict__ attn, float* __restrict__ col)
{
  const int t = blockIdx.x * 256 + threadIdx.x;   // < 112896
  const int gid = t / 9, jj = t % 9;
  const int bg = gid / 784, pos = gid % 784;
  const int bb = bg >> 2;
  const int i = pos / 28, j = pos % 28;
  const int si = min(max(i - 1, 0), 25), sj = min(max(j - 1, 0), 25);
  const int p = (si + jj / 3) * 28 + (sj + jj % 3);
  atomicAdd(&col[bb * 784 + p], attn[t]);
}

// AV + residual: x_out[b,i,j,:] += sum_g sum_{a,c} (attn/denom) * v[b,g,ih,iw,:]
__global__ __launch_bounds__(256) void na_group_av(
    const float* __restrict__ attn, const float* __restrict__ col,
    const float* __restrict__ Vg, float* __restrict__ Xout)
{
  const int wv = threadIdx.x >> 5, lane = threadIdx.x & 31;
  const int wid = blockIdx.x * 8 + wv;            // b*784 + pos
  const int bb = wid / 784, pos = wid % 784;
  const int i = pos / 28, j = pos % 28;
  const int si = min(max(i - 1, 0), 25), sj = min(max(j - 1, 0), 25);
  float acc[8] = {0.f,0.f,0.f,0.f,0.f,0.f,0.f,0.f};
  for (int g = 0; g < 4; ++g) {
    const size_t gid = ((size_t)(bb * 4 + g)) * 784 + pos;
    const float* ap = attn + gid * 9;
#pragma unroll
    for (int a = 0; a < 3; ++a)
#pragma unroll
      for (int c = 0; c < 3; ++c) {
        const int p = (si + a) * 28 + (sj + c);
        const float av = ap[a * 3 + c] / (col[bb * 784 + p] + 1e-8f);
        const float* vp = Vg + (((size_t)(bb * 4 + g)) * 784 + p) * 256 + lane * 8;
#pragma unroll
        for (int e = 0; e < 8; ++e) acc[e] += av * vp[e];
      }
  }
  float* xo = Xout + (size_t)wid * 256 + lane * 8;
#pragma unroll
  for (int e = 0; e < 8; ++e) xo[e] += acc[e];
}

// ---------------------------------------------------------------------------
// RoPE in-place on q and k slices of qkv buffer (row = [q 256 | k 256 | v 256]).
// ---------------------------------------------------------------------------
__global__ void rope_qk(float* __restrict__ qkv)
{
  const int t = blockIdx.x * 256 + threadIdx.x;   // < 802816 = 3136*256
  const int token = t >> 8, r = t & 255;
  const int qk = r >> 7, hp = r & 127;
  const int head = hp >> 4, i2 = hp & 15;
  const int n = token % 784;
  const float inv = expf(-(float)(2 * i2) * (1.f / 32.f) * 9.21034037197618273607f); // ln(10000)
  const float ang = (float)n * inv;
  float s, c;
  sincosf(ang, &s, &c);
  const size_t base = (size_t)token * 768 + qk * 256 + head * 32 + i2 * 2;
  const float x0 = qkv[base], x1 = qkv[base + 1];
  qkv[base]     = x0 * c - x1 * s;
  qkv[base + 1] = x1 * c + x0 * s;
}

// Block NA (7x7), one wave per (b,h,i,j); lane = head dim (HD=32).
__global__ __launch_bounds__(256) void na_head_qk(
    const float* __restrict__ qkv, float* __restrict__ attn)
{
  const int wv = threadIdx.x >> 5, lane = threadIdx.x & 31;
  const int wid = blockIdx.x * 8 + wv;            // (b*8+h)*784 + pos
  const int bh = wid / 784, pos = wid % 784;
  const int bb = bh >> 3, h = bh & 7;
  const int i = pos / 28, j = pos % 28;
  const int si = min(max(i - 3, 0), 21), sj = min(max(j - 3, 0), 21);
  const size_t trow = (size_t)bb * 784 + pos;
  const float qd = qkv[trow * 768 + h * 32 + lane];
  float* ap = attn + (size_t)wid * 49;
  for (int a = 0; a < 7; ++a)
#pragma unroll
    for (int c = 0; c < 7; ++c) {
      const size_t nt = (size_t)bb * 784 + (si + a) * 28 + (sj + c);
      const float red = wave_sum(qd * qkv[nt * 768 + 256 + h * 32 + lane]);
      if (lane == 0) ap[a * 7 + c] = red * 0.17677669529663688110f;  // 32^-0.5
    }
}

__global__ __launch_bounds__(256) void softmax49(float* __restrict__ attn)
{
  const int wv = threadIdx.x >> 5, lane = threadIdx.x & 31;
  const size_t wid = (size_t)blockIdx.x * 8 + wv;
  float* ap = attn + wid * 49;
  const float v0 = (lane < 49)      ? ap[lane]      : -3e38f;
  const float v1 = (lane + 32 < 49) ? ap[lane + 32] : -3e38f;
  const float m = wave_max(fmaxf(v0, v1));
  const float e0 = (lane < 49)      ? expf(v0 - m) : 0.f;
  const float e1 = (lane + 32 < 49) ? expf(v1 - m) : 0.f;
  const float inv = 1.f / wave_sum(e0 + e1);
  if (lane < 49)      ap[lane]      = e0 * inv;
  if (lane + 32 < 49) ap[lane + 32] = e1 * inv;
}

__global__ __launch_bounds__(256) void na_head_av(
    const float* __restrict__ qkv, const float* __restrict__ attn,
    float* __restrict__ obuf)
{
  const int wv = threadIdx.x >> 5, lane = threadIdx.x & 31;
  const int wid = blockIdx.x * 8 + wv;
  const int bh = wid / 784, pos = wid % 784;
  const int bb = bh >> 3, h = bh & 7;
  const int i = pos / 28, j = pos % 28;
  const int si = min(max(i - 3, 0), 21), sj = min(max(j - 3, 0), 21);
  const float* ap = attn + (size_t)wid * 49;
  float acc = 0.f;
  for (int a = 0; a < 7; ++a)
#pragma unroll
    for (int c = 0; c < 7; ++c) {
      const size_t nt = (size_t)bb * 784 + (si + a) * 28 + (sj + c);
      acc += ap[a * 7 + c] * qkv[nt * 768 + 512 + h * 32 + lane];
    }
  obuf[((size_t)bb * 784 + pos) * 256 + h * 32 + lane] = acc;
}

// ---------------------------------------------------------------------------
extern "C" void kernel_launch(void* const* d_in, const int* in_sizes, int n_in,
                              void* d_out, int out_size, void* d_ws, size_t ws_size,
                              hipStream_t stream)
{
  (void)in_sizes; (void)n_in; (void)out_size; (void)ws_size;
  const float* x          = (const float*)d_in[0];
  const float* g_seed_w   = (const float*)d_in[1];
  const float* g_q_w      = (const float*)d_in[2];
  const float* g_k_w      = (const float*)d_in[3];
  const float* g_v_w      = (const float*)d_in[4];
  const float* g_mlp_w1   = (const float*)d_in[5];
  const float* g_mlp_b1   = (const float*)d_in[6];
  const float* g_mlp_w2   = (const float*)d_in[7];
  const float* g_mlp_b2   = (const float*)d_in[8];
  const float* g_ln_in_g  = (const float*)d_in[9];
  const float* g_ln_in_b  = (const float*)d_in[10];
  const float* g_ln_out_g = (const float*)d_in[11];
  const float* g_ln_out_b = (const float*)d_in[12];
  const float* g_tau      = (const float*)d_in[13];
  const float* g_rpb      = (const float*)d_in[14];
  const float* blk_ln1_g  = (const float*)d_in[15];
  const float* blk_ln1_b  = (const float*)d_in[16];
  const float* blk_qkv_w  = (const float*)d_in[17];
  const float* blk_proj_w = (const float*)d_in[18];
  const float* blk_proj_b = (const float*)d_in[19];
  const float* blk_ln2_g  = (const float*)d_in[20];
  const float* blk_ln2_b  = (const float*)d_in[21];
  const float* blk_mlp_w1 = (const float*)d_in[22];
  const float* blk_mlp_b1 = (const float*)d_in[23];
  const float* blk_mlp_w2 = (const float*)d_in[24];
  const float* blk_mlp_b2 = (const float*)d_in[25];

  float* xout = (float*)d_out;                 // (4, 784, 256) — final result
  float* ws = (float*)d_ws;
  size_t off = 0;
  float* kg   = ws + off; off += 3211264;      // (B,4,28,28,256)
  float* vg   = ws + off; off += 3211264;
  float* lnx  = ws + off; off += 3211264;      // ln_in(x)
  float* tmp  = ws + off; off += 802816;       // ln scratch (L,256)
  float* qkv  = ws + off; off += 2408448;      // q (L,256) / qkv (L,768)
  float* h1   = ws + off; off += 2408448;      // MLP hidden (L,<=768)
  float* h2   = ws + off; off += 802816;       // MLP out / seed
  float* attn = ws + off; off += 1229312;      // max(B*4*784*9, B*8*784*49)
  float* col  = ws + off; off += 3136;         // (B,784)
  float* obuf = ws + off; off += 802816;       // attention output (L,256)

  const dim3 blk(256);

  // --- k, v projections (directly into grouped layout), seed conv + ln ---
  ln256<<<1568, blk, 0, stream>>>(x, g_ln_in_g, g_ln_in_b, lnx, 0);
  gemm_wmma<GF_PERM><<<dim3(2, 196), blk, 0, stream>>>(lnx, 256, g_k_w, 256, nullptr, kg, 256, 256);
  gemm_wmma<GF_PERM><<<dim3(2, 196), blk, 0, stream>>>(x,   256, g_v_w, 256, nullptr, vg, 256, 256);
  conv_seed_wmma<<<dim3(4, 49), blk, 0, stream>>>(x, g_seed_w, h2);
  ln256<<<392, blk, 0, stream>>>(h2, g_ln_out_g, g_ln_out_b, xout, 0);

  // --- 3 iterations: q-proj, 3x3 NA with column renorm, MLP + ln residual ---
  for (int it = 0; it < 3; ++it) {
    ln256<<<392, blk, 0, stream>>>(xout, g_ln_out_g, g_ln_out_b, tmp, 0);
    gemm_wmma<0><<<dim3(2, 49), blk, 0, stream>>>(tmp, 256, g_q_w, 256, nullptr, qkv, 256, 256);
    zero_f32<<<13, blk, 0, stream>>>(col, 3136);
    na_group_qk<<<1568, blk, 0, stream>>>(qkv, kg, g_rpb, g_tau, attn);
    na_group_scatter<<<441, blk, 0, stream>>>(attn, col);
    na_group_av<<<392, blk, 0, stream>>>(attn, col, vg, xout);
    gemm_wmma<GF_BIAS | GF_GELU><<<dim3(4, 49), blk, 0, stream>>>(
        xout, 256, g_mlp_w1, 256, g_mlp_b1, h1, 512, 256);
    gemm_wmma<GF_BIAS><<<dim3(2, 49), blk, 0, stream>>>(
        h1, 512, g_mlp_w2, 512, g_mlp_b2, h2, 256, 512);
    ln256<<<392, blk, 0, stream>>>(h2, g_ln_out_g, g_ln_out_b, xout, 1);
  }

  // --- 2 transformer blocks: qkv + RoPE + 7x7 NA + proj + MLP ---
  for (int l = 0; l < 2; ++l) {
    ln256<<<392, blk, 0, stream>>>(xout, blk_ln1_g + l * 256, blk_ln1_b + l * 256, tmp, 0);
    gemm_wmma<0><<<dim3(6, 49), blk, 0, stream>>>(
        tmp, 256, blk_qkv_w + (size_t)l * 768 * 256, 256, nullptr, qkv, 768, 256);
    rope_qk<<<3136, blk, 0, stream>>>(qkv);
    na_head_qk<<<3136, blk, 0, stream>>>(qkv, attn);
    softmax49<<<3136, blk, 0, stream>>>(attn);
    na_head_av<<<3136, blk, 0, stream>>>(qkv, attn, obuf);
    gemm_wmma<GF_BIAS | GF_RESID><<<dim3(2, 49), blk, 0, stream>>>(
        obuf, 256, blk_proj_w + (size_t)l * 256 * 256, 256, blk_proj_b + l * 256, xout, 256, 256);
    ln256<<<392, blk, 0, stream>>>(xout, blk_ln2_g + l * 256, blk_ln2_b + l * 256, tmp, 0);
    gemm_wmma<GF_BIAS | GF_GELU><<<dim3(6, 49), blk, 0, stream>>>(
        tmp, 256, blk_mlp_w1 + (size_t)l * 768 * 256, 256, blk_mlp_b1 + l * 768, h1, 768, 256);
    gemm_wmma<GF_BIAS | GF_RESID><<<dim3(2, 49), blk, 0, stream>>>(
        h1, 768, blk_mlp_w2 + (size_t)l * 256 * 768, 768, blk_mlp_b2 + l * 256, xout, 256, 768);
  }
}
